// Indexer_48447231098919
// MI455X (gfx1250) — matrix operand order repaired and checked by reading, not compiled, not run on hardware
//
#include <hip/hip_runtime.h>

typedef __bf16 bf16;
typedef __bf16 v16bf __attribute__((ext_vector_type(16)));
typedef float  v8f   __attribute__((ext_vector_type(8)));
typedef unsigned int v4u __attribute__((ext_vector_type(4)));
typedef int v4i __attribute__((vector_size(16)));

#define T_    2048
#define EMB_  2048
#define NH_   32
#define HD_   128
#define QL_   1536
#define TOPK_ 512
#define NDIM_ (NH_ * HD_)  /* 4096 */
#define MASK_VAL (-0.7f * 3.40282346638528859812e38f)
#define LOG1E4_OVER_32 0.28782313662425572f /* ln(10000)/32 */

#if defined(__has_builtin)
#if __has_builtin(__builtin_amdgcn_global_load_async_to_lds_b128)
#define HAVE_ASYNC_LDS 1
#endif
#endif

#ifdef HAVE_ASYNC_LDS
#pragma message("gfx1250: using GLOBAL_LOAD_ASYNC_TO_LDS_B128 path")
#else
#pragma message("gfx1250: async-to-LDS builtin NOT found; using sync LDS copy fallback")
#endif

union FragB16 { v16bf v; v4u u[2]; };

// Copy 16B global -> LDS. Async (ASYNCcnt-tracked) when available.
__device__ __forceinline__ void cp16_g2l(const bf16* g, bf16* l) {
#ifdef HAVE_ASYNC_LDS
  __builtin_amdgcn_global_load_async_to_lds_b128((v4i*)g, (v4i*)l, 0, 0);
#else
  *(v4u*)l = *(const v4u*)g;
#endif
}

__device__ __forceinline__ void cp_wait_all() {
#ifdef HAVE_ASYNC_LDS
#if __has_builtin(__builtin_amdgcn_s_wait_asynccnt)
  __builtin_amdgcn_s_wait_asynccnt(0);
#else
  asm volatile("s_wait_asynccnt 0x0" ::: "memory");
#endif
#endif
}

// B-matrix 32x16 bf16 fragment: lanes 0-15 -> N=lane, K=kb+0..7,16..23; lanes 16-31 -> N=lane-16, K=kb+8..15,24..31.
// Source stored N-major (row n holds K contiguously), row stride ld.
__device__ __forceinline__ v16bf load_frag_b(const bf16* __restrict__ srcT, int n0, int kb, int ld) {
  int lane = threadIdx.x & 31;
  int n    = n0 + (lane & 15);
  int klo  = kb + ((lane & 16) ? 8 : 0);
  const bf16* p = srcT + (long)n * ld + klo;
  FragB16 f;
  f.u[0] = *(const v4u*)(p);
  f.u[1] = *(const v4u*)(p + 16);
  return f.v;
}

// ---- bf16 converts -------------------------------------------------------
__global__ void cvt_lrq_kernel(const float* __restrict__ x, bf16* __restrict__ y, int n) {
  int i = blockIdx.x * blockDim.x + threadIdx.x;
  if (i < n) y[i] = (bf16)x[i];
}

// wq_b [l][n] (n = h*128+d) -> transposed bf16 [n][l] so B-fragments load contiguously.
__global__ void cvt_wqb_kernel(const float* __restrict__ w, bf16* __restrict__ yT) {
  int i = blockIdx.x * blockDim.x + threadIdx.x;
  if (i < QL_ * NDIM_) {
    int l = i >> 12;        // / 4096
    int n = i & (NDIM_ - 1);
    yT[(long)n * QL_ + l] = (bf16)w[i];
  }
}

// ---- k = rope(LN(kv @ wk)) -> bf16 [s][128] ------------------------------
__global__ __launch_bounds__(128) void k_kernel(const float* __restrict__ kv, const float* __restrict__ wk,
                                                const float* __restrict__ lns, const float* __restrict__ lnb,
                                                const int* __restrict__ pos, bf16* __restrict__ kout) {
  int s = blockIdx.x, d = threadIdx.x;
  const float* kvr = kv + (long)s * EMB_;
  float acc = 0.f;
  for (int e = 0; e < EMB_; ++e) acc = fmaf(kvr[e], wk[(long)e * HD_ + d], acc);

  __shared__ float buf[HD_];
  __shared__ float stats[2];
  buf[d] = acc;
  __syncthreads();
  if (d == 0) {
    float su = 0.f;
    for (int i = 0; i < HD_; ++i) su += buf[i];
    float mu = su * (1.0f / HD_);
    float var = 0.f;
    for (int i = 0; i < HD_; ++i) { float x = buf[i] - mu; var += x * x; }
    stats[0] = mu;
    stats[1] = rsqrtf(var * (1.0f / HD_) + 1e-6f);
  }
  __syncthreads();
  float x = (acc - stats[0]) * stats[1] * lns[d] + lnb[d];
  buf[d] = x;
  __syncthreads();

  float p = (float)pos[s];
  float out;
  if (d < 32) {
    float th = p * __expf(-(float)d * LOG1E4_OVER_32);
    out = buf[d] * __cosf(th) - buf[d + 32] * __sinf(th);
  } else if (d < 64) {
    int i = d - 32;
    float th = p * __expf(-(float)i * LOG1E4_OVER_32);
    out = buf[d] * __cosf(th) + buf[i] * __sinf(th);
  } else {
    out = buf[d];
  }
  kout[(long)s * HD_ + d] = (bf16)out;
}

// ---- weightsT[h][t] = (x_q[t]·w_proj[:,h]) / 64 --------------------------
__global__ void w_kernel(const float* __restrict__ xq, const float* __restrict__ wp, float* __restrict__ wT) {
  int t = blockIdx.x, h = threadIdx.x;
  const float* xr = xq + (long)t * EMB_;
  float acc = 0.f;
  for (int e = 0; e < EMB_; ++e) acc = fmaf(xr[e], wp[(long)e * NH_ + h], acc);
  wT[(long)h * T_ + t] = acc * 0.015625f;  // NH^-.5 * HD^-.5 = 1/64
}

// ---- Q GEMM: q[t][n] = lrq[t][:] . wq_b[:][n], bf16 WMMA, K=1536 ---------
// 8 waves/block share one 16-row lrq tile staged in LDS via async copies.
__global__ __launch_bounds__(256) void qgemm_kernel(const bf16* __restrict__ lrq, const bf16* __restrict__ wqT,
                                                    bf16* __restrict__ qout) {
  __shared__ __align__(16) bf16 atile[16 * QL_];  // 48KB
  int tid = threadIdx.x;
  int wave = tid >> 5, lane = tid & 31;
  int t0 = blockIdx.x * 16;
  int n0 = (blockIdx.y * 8 + wave) * 16;

  // stage 16 x 1536 bf16 (3072 x 16B, 12 per thread)
  const bf16* src = lrq + (long)t0 * QL_;
#pragma unroll
  for (int it = 0; it < 12; ++it) {
    int i = it * 256 + tid;
    int r = i / 192;              // 192 = 1536/8 16B-chunks per row
    int c = (i % 192) * 8;
    cp16_g2l(src + (long)r * QL_ + c, &atile[r * QL_ + c]);
  }
  cp_wait_all();
  __syncthreads();

  int arow = lane & 15;
  int aoff = (lane & 16) ? 8 : 0;   // A layout: lanes16-31 take K+8 / K+24
  v8f acc = {0.f, 0.f, 0.f, 0.f, 0.f, 0.f, 0.f, 0.f};
  for (int kb = 0; kb < QL_; kb += 32) {
    const bf16* ap = &atile[arow * QL_ + kb + aoff];
    FragB16 fa;
    fa.u[0] = *(const v4u*)(ap);
    fa.u[1] = *(const v4u*)(ap + 16);
    v16bf b = load_frag_b(wqT, n0, kb, QL_);
    acc = __builtin_amdgcn_wmma_f32_16x16x32_bf16(false, fa.v, false, b, (short)0, acc, false, false);
  }
  int n = n0 + (lane & 15);
  int mbase = t0 + ((lane & 16) ? 8 : 0);
#pragma unroll
  for (int r = 0; r < 8; ++r) qout[(long)(mbase + r) * NDIM_ + n] = (bf16)acc[r];
}

// ---- in-place partial rope on q bf16 [t][h][128], pairs (i, i+32), i<32 --
__global__ void ropeq_kernel(bf16* __restrict__ q, const int* __restrict__ pos) {
  int idx = blockIdx.x * blockDim.x + threadIdx.x;
  if (idx >= T_ * NH_ * 32) return;
  int t = idx >> 10;
  int r = idx & 1023;
  int h = r >> 5, i = r & 31;
  long base = (long)t * NDIM_ + (long)h * HD_;
  float a = (float)q[base + i], b = (float)q[base + i + 32];
  float th = (float)pos[t] * __expf(-(float)i * LOG1E4_OVER_32);
  float c = __cosf(th), sn = __sinf(th);
  q[base + i]      = (bf16)(a * c - b * sn);
  q[base + i + 32] = (bf16)(b * c + a * sn);
}

// ---- fused score: score[t][s] = sum_h relu(q[t,h,:]·k[s,:]) * w[t,h] -----
// Block: 8 waves, one 16x16 (t,s) tile per wave, shared t0. q t-tile staged
// in LDS in two 16-head phases (64KB); weights staged in LDS (2KB).
__global__ __launch_bounds__(256) void score_kernel(const bf16* __restrict__ q, const bf16* __restrict__ k,
                                                    const float* __restrict__ wT, float* __restrict__ score) {
  __shared__ __align__(16) bf16  qtile[16 * 2048];  // 64KB: 16 rows x 16 heads x 128
  __shared__ __align__(16) float wtile[NH_ * 16];   // 2KB
  int tid = threadIdx.x;
  int wave = tid >> 5, lane = tid & 31;
  int t0 = blockIdx.x * 16;
  int s0 = (blockIdx.y * 8 + wave) * 16;

  for (int i = tid; i < NH_ * 16; i += 256) {
    int h = i >> 4, m = i & 15;
    wtile[i] = wT[(long)h * T_ + t0 + m];
  }

  v16bf bk0 = load_frag_b(k, s0, 0,  HD_);
  v16bf bk1 = load_frag_b(k, s0, 32, HD_);
  v16bf bk2 = load_frag_b(k, s0, 64, HD_);
  v16bf bk3 = load_frag_b(k, s0, 96, HD_);

  v8f acc = {0.f, 0.f, 0.f, 0.f, 0.f, 0.f, 0.f, 0.f};
  int mbase = (lane & 16) ? 8 : 0;
  int arow  = lane & 15;
  int aoff  = (lane & 16) ? 8 : 0;

  for (int p = 0; p < 2; ++p) {
    __syncthreads();  // previous phase fully consumed (and wtile visible)
    // stage 16 rows x 2048 bf16 (4096 x 16B, 16 per thread)
    const bf16* src = q + (long)t0 * NDIM_ + p * 2048;
#pragma unroll
    for (int it = 0; it < 16; ++it) {
      int i = it * 256 + tid;
      int r = i >> 8;           // 256 16B-chunks per row
      int c = (i & 255) << 3;
      cp16_g2l(src + (long)r * NDIM_ + c, &qtile[r * 2048 + c]);
    }
    cp_wait_all();
    __syncthreads();

    for (int hh = 0; hh < 16; ++hh) {
      int h = p * 16 + hh;
      const bf16* qb = &qtile[arow * 2048 + hh * HD_ + aoff];
      v8f l = {0.f, 0.f, 0.f, 0.f, 0.f, 0.f, 0.f, 0.f};
      FragB16 fa;
      fa.u[0] = *(const v4u*)(qb);       fa.u[1] = *(const v4u*)(qb + 16);
      l = __builtin_amdgcn_wmma_f32_16x16x32_bf16(false, fa.v, false, bk0, (short)0, l, false, false);
      fa.u[0] = *(const v4u*)(qb + 32);  fa.u[1] = *(const v4u*)(qb + 48);
      l = __builtin_amdgcn_wmma_f32_16x16x32_bf16(false, fa.v, false, bk1, (short)0, l, false, false);
      fa.u[0] = *(const v4u*)(qb + 64);  fa.u[1] = *(const v4u*)(qb + 80);
      l = __builtin_amdgcn_wmma_f32_16x16x32_bf16(false, fa.v, false, bk2, (short)0, l, false, false);
      fa.u[0] = *(const v4u*)(qb + 96);  fa.u[1] = *(const v4u*)(qb + 112);
      l = __builtin_amdgcn_wmma_f32_16x16x32_bf16(false, fa.v, false, bk3, (short)0, l, false, false);

      const float4* wp4 = (const float4*)(&wtile[h * 16 + mbase]);
      float4 wa = wp4[0], wb = wp4[1];
      acc[0] += fmaxf(l[0], 0.f) * wa.x;
      acc[1] += fmaxf(l[1], 0.f) * wa.y;
      acc[2] += fmaxf(l[2], 0.f) * wa.z;
      acc[3] += fmaxf(l[3], 0.f) * wa.w;
      acc[4] += fmaxf(l[4], 0.f) * wb.x;
      acc[5] += fmaxf(l[5], 0.f) * wb.y;
      acc[6] += fmaxf(l[6], 0.f) * wb.z;
      acc[7] += fmaxf(l[7], 0.f) * wb.w;
    }
  }

  int n = s0 + (lane & 15);
#pragma unroll
  for (int r = 0; r < 8; ++r) score[(long)(t0 + mbase + r) * T_ + n] = acc[r];
}

// ---- top-512 per row via radix-select on order-preserving uint keys ------
__global__ __launch_bounds__(256) void topk_kernel(const float* __restrict__ score, float* __restrict__ mask,
                                                   int* __restrict__ idxout) {
  int t = blockIdx.x, tid = threadIdx.x;
  __shared__ unsigned keys[T_];
  __shared__ int cnt;

  const float* row = score + (long)t * T_;
  for (int j = tid; j < T_; j += 256) {
    unsigned u = __float_as_uint(row[j]);
    keys[j] = (u & 0x80000000u) ? ~u : (u | 0x80000000u);
  }
  __syncthreads();

  unsigned prefix = 0;
  for (int bit = 31; bit >= 0; --bit) {
    unsigned cand = prefix | (1u << bit);
    if (tid == 0) cnt = 0;
    __syncthreads();
    int local = 0;
    for (int j = tid; j < T_; j += 256) local += (keys[j] >= cand) ? 1 : 0;
    if (local) atomicAdd(&cnt, local);
    __syncthreads();
    if (cnt >= TOPK_) prefix = cand;  // uniform across block
    __syncthreads();
  }

  // prefix == 512th-largest key. Select all > prefix, then == prefix until 512.
  if (tid == 0) cnt = 0;
  __syncthreads();
  float* mrow = mask + (long)t * T_;
  int* irow = idxout + t * TOPK_;
  for (int j = tid; j < T_; j += 256) {
    unsigned kk = keys[j];
    if (kk > prefix) {
      int p = atomicAdd(&cnt, 1);
      irow[p] = j;
      mrow[j] = 0.f;
    } else if (kk < prefix) {
      mrow[j] = MASK_VAL;
    }
  }
  __syncthreads();
  for (int j = tid; j < T_; j += 256) {
    if (keys[j] == prefix) {
      int p = atomicAdd(&cnt, 1);
      if (p < TOPK_) { irow[p] = j; mrow[j] = 0.f; }
      else           { mrow[j] = MASK_VAL; }
    }
  }
}

extern "C" void kernel_launch(void* const* d_in, const int* in_sizes, int n_in,
                              void* d_out, int out_size, void* d_ws, size_t ws_size,
                              hipStream_t stream) {
  const float* inputs_q   = (const float*)d_in[0];
  const float* low_rank_q = (const float*)d_in[1];
  const float* inputs_kv  = (const float*)d_in[2];
  const int*   pos        = (const int*)d_in[3];
  const float* wq_b       = (const float*)d_in[4];
  const float* wk         = (const float*)d_in[5];
  const float* w_proj     = (const float*)d_in[6];
  const float* ln_scale   = (const float*)d_in[7];
  const float* ln_bias    = (const float*)d_in[8];

  char* ws = (char*)d_ws;
  bf16* lrq_b = (bf16*)ws;  ws += (size_t)T_ * QL_ * 2;      // 6.3 MB
  bf16* wqT_b = (bf16*)ws;  ws += (size_t)NDIM_ * QL_ * 2;   // 12.6 MB
  bf16* q_b   = (bf16*)ws;  ws += (size_t)T_ * NDIM_ * 2;    // 16.8 MB
  bf16* k_b   = (bf16*)ws;  ws += (size_t)T_ * HD_ * 2;      // 0.5 MB
  float* wT   = (float*)ws; ws += (size_t)NH_ * T_ * 4;      // 0.25 MB

  // outputs: mask [T,T] f32 | topk_indices [T,512] i32 | index_score [T,T] f32
  float* mask  = (float*)d_out;
  int*   tkidx = (int*)((float*)d_out + (size_t)T_ * T_);
  float* score = (float*)d_out + (size_t)T_ * T_ + (size_t)T_ * TOPK_;

  cvt_lrq_kernel<<<(T_ * QL_ + 255) / 256, 256, 0, stream>>>(low_rank_q, lrq_b, T_ * QL_);
  cvt_wqb_kernel<<<(QL_ * NDIM_ + 255) / 256, 256, 0, stream>>>(wq_b, wqT_b);
  k_kernel<<<T_, HD_, 0, stream>>>(inputs_kv, wk, ln_scale, ln_bias, pos, k_b);
  w_kernel<<<T_, NH_, 0, stream>>>(inputs_q, w_proj, wT);
  qgemm_kernel<<<dim3(T_ / 16, NDIM_ / 16 / 8), 256, 0, stream>>>(lrq_b, wqT_b, q_b);
  ropeq_kernel<<<(T_ * NH_ * 32 + 255) / 256, 256, 0, stream>>>(q_b, pos);
  score_kernel<<<dim3(T_ / 16, T_ / 16 / 8), 256, 0, stream>>>(q_b, k_b, wT, score);
  topk_kernel<<<T_, 256, 0, stream>>>(score, mask, tkidx);
}
